// PagedTransformerBlock_78460462563936
// MI455X (gfx1250) — compile-verified, matrix operand order checked by reading, not run on hardware
//
#include <hip/hip_runtime.h>
#include <hip/hip_bf16.h>

// ---- problem constants (fixed by setup_inputs) ----
#define DIM    4096
#define NHEAD  32
#define HDSZ   128                 // head dim
#define HDIM   (NHEAD*HDSZ)        // 4096
#define QKV_LD (3*HDIM)            // 12288
#define FF     16384               // half of gate_up output
#define TPF    2048                // prefill tokens
#define NBD    8                   // decode batch
#define NTOK   (TPF+NBD)           // 2056
#define MTILES ((NTOK+127)/128)    // 17
#define MPAD   (MTILES*128)        // 2176 (padded rows for unguarded stores)
#define MAXB_  64
#define BLOCK_ 16
#define PB_    2
#define EPSV   1e-5f
#define NEGV   (-1e30f)

#define LDS_STRIDE 20              // dwords per 32-bf16 row: 16B aligned, bank-conflict-free

typedef __attribute__((ext_vector_type(16))) __bf16    bf16x16;
typedef __attribute__((ext_vector_type(8)))  float     f32x8;
typedef __attribute__((ext_vector_type(4)))  unsigned  u32x4;
typedef __attribute__((ext_vector_type(2)))  unsigned  u32x2;

union FragBF { bf16x16 v; u32x4 q[2]; };

__device__ __forceinline__ unsigned short f2bf(float f) {
  unsigned u = __float_as_uint(f);
  unsigned r = u + 0x7fffu + ((u >> 16) & 1u);   // round-to-nearest-even
  return (unsigned short)(r >> 16);
}
__device__ __forceinline__ unsigned pack2(float a, float b) {
  return (unsigned)f2bf(a) | ((unsigned)f2bf(b) << 16);
}
__device__ __forceinline__ float wred_max(float v) {
#pragma unroll
  for (int o = 16; o > 0; o >>= 1) v = fmaxf(v, __shfl_xor(v, o, 32));
  return v;
}
__device__ __forceinline__ float wred_sum(float v) {
#pragma unroll
  for (int o = 16; o > 0; o >>= 1) v += __shfl_xor(v, o, 32);
  return v;
}

// ---------------- RMSNorm -> bf16 ----------------
__global__ __launch_bounds__(256) void rmsnorm_bf16_kernel(
    const float* __restrict__ x, const float* __restrict__ w,
    unsigned short* __restrict__ out) {
  int row = blockIdx.x;
  const float4* xr = reinterpret_cast<const float4*>(x + (size_t)row * DIM);
  const float4* w4 = reinterpret_cast<const float4*>(w);
  float ss = 0.f;
  float4 vals[4];
#pragma unroll
  for (int i = 0; i < 4; i++) {
    float4 v = xr[threadIdx.x + 256 * i];
    vals[i] = v;
    ss += v.x * v.x + v.y * v.y + v.z * v.z + v.w * v.w;
  }
  ss = wred_sum(ss);
  __shared__ float red[8];
  if ((threadIdx.x & 31) == 0) red[threadIdx.x >> 5] = ss;
  __syncthreads();
  float tot = 0.f;
#pragma unroll
  for (int i = 0; i < 8; i++) tot += red[i];
  float rs = rsqrtf(tot / (float)DIM + EPSV);
  unsigned* o32 = reinterpret_cast<unsigned*>(out) + (size_t)row * (DIM / 2);
#pragma unroll
  for (int i = 0; i < 4; i++) {
    int idx = threadIdx.x + 256 * i;
    float4 v = vals[i];
    float4 g = w4[idx];
    o32[idx * 2]     = pack2(v.x * rs * g.x, v.y * rs * g.y);
    o32[idx * 2 + 1] = pack2(v.z * rs * g.z, v.w * rs * g.w);
  }
}

// ---- GEMM epilogue (compile-time N; optional per-row guard) ----
template <int MODE, int N, bool CHECK>
__device__ __forceinline__ void gemm_store(
    const f32x8 (&acc)[4][2], const f32x8 (&acc2)[4][2],
    float* __restrict__ outF, unsigned short* __restrict__ outBF,
    const float* __restrict__ resid,
    int mbase, int nbase, int wm, int wn, int lane) {
#pragma unroll
  for (int t = 0; t < 4; t++) {
    int m0 = mbase + wm * 64 + t * 16 + ((lane >= 16) ? 8 : 0);
#pragma unroll
    for (int s = 0; s < 2; s++) {
      int n = nbase + wn * 32 + s * 16 + (lane & 15);
#pragma unroll
      for (int r = 0; r < 8; r++) {
        int m = m0 + r;
        if (CHECK && m >= NTOK) continue;
        float v = acc[t][s][r];
        if (MODE == 0) {
          outF[(size_t)m * N + n] = v;
        } else if (MODE == 1) {
          outF[(size_t)m * N + n] = v + resid[(size_t)m * N + n];
        } else {
          float u = acc2[t][s][r];
          float si = v / (1.f + __expf(-v));
          outBF[(size_t)m * N + n] = f2bf(si * u);
        }
      }
    }
  }
}

// ---------------- WMMA GEMM: C[M,N] = A[M,K](bf16) * W[N,K]^T ----------------
// MODE 0: write f32 out               (out buffer padded to MPAD rows)
// MODE 1: write f32 out + residual    (padded unless GUARD)
// MODE 2: gate/up fused: u = A*W[N+n]^T, writes bf16 silu(g)*u (padded)
// GUARD : true only when the output buffer is exactly NTOK rows (d_out)
template <int MODE, int N, int K, bool GUARD>
__global__ __launch_bounds__(256) void gemm_wmma_kernel(
    const unsigned short* __restrict__ Abf, const float* __restrict__ W,
    float* __restrict__ outF, unsigned short* __restrict__ outBF,
    const float* __restrict__ resid) {
  __shared__ __align__(16) unsigned Ash[128 * LDS_STRIDE];
  __shared__ __align__(16) unsigned Bsh[128 * LDS_STRIDE];
  __shared__ __align__(16) unsigned Bsh2[(MODE == 2) ? 128 * LDS_STRIDE : 4];

  const int tid = threadIdx.x;
  const int lane = tid & 31;
  const int wid = tid >> 5;
  const int wm = wid & 1;   // 2 waves along M
  const int wn = wid >> 1;  // 4 waves along N
  const int mbase = blockIdx.x * 128;
  const int nbase = blockIdx.y * 128;
  constexpr int Kh = K >> 1;  // dwords per A row
  const unsigned long long Abase = (unsigned long long)(uintptr_t)Abf;

  f32x8 zero = {0.f, 0.f, 0.f, 0.f, 0.f, 0.f, 0.f, 0.f};
  f32x8 acc[4][2];
  f32x8 acc2[4][2];
#pragma unroll
  for (int t = 0; t < 4; t++)
#pragma unroll
    for (int s = 0; s < 2; s++) { acc[t][s] = zero; acc2[t][s] = zero; }

  // per-thread staging coordinates
  // A tile: 128 rows x 32 bf16 = 512 b128 chunks; 2 per thread (async to LDS)
  const int arow0 = (tid + 0)   >> 2, ac0 = (tid + 0)   & 3;
  const int arow1 = (tid + 256) >> 2, ac1 = (tid + 256) & 3;
  int am0 = mbase + arow0; if (am0 >= NTOK) am0 = NTOK - 1;  // clamp (stores guarded/padded)
  int am1 = mbase + arow1; if (am1 >= NTOK) am1 = NTOK - 1;
  const unsigned alds0 = (unsigned)(uintptr_t)&Ash[arow0 * LDS_STRIDE + ac0 * 4];
  const unsigned alds1 = (unsigned)(uintptr_t)&Ash[arow1 * LDS_STRIDE + ac1 * 4];

  for (int k0 = 0; k0 < K; k0 += 32) {
    // ---- stage A tile via CDNA5 async global->LDS (b128 per lane) ----
    {
      unsigned voff0 = (unsigned)(((size_t)am0 * Kh + (k0 >> 1) + ac0 * 4) * 4u);
      unsigned voff1 = (unsigned)(((size_t)am1 * Kh + (k0 >> 1) + ac1 * 4) * 4u);
      asm volatile("global_load_async_to_lds_b128 %0, %1, %2"
                   :: "v"(alds0), "v"(voff0), "s"(Abase) : "memory");
      asm volatile("global_load_async_to_lds_b128 %0, %1, %2"
                   :: "v"(alds1), "v"(voff1), "s"(Abase) : "memory");
    }
    // ---- stage W tile(s) 128x32 f32 -> bf16 (needs conversion; sync path) ----
#pragma unroll
    for (int i = 0; i < 4; i++) {
      int lin = tid + 256 * i;
      int row = lin >> 3, c4 = lin & 7;
      const float4 wv = *reinterpret_cast<const float4*>(
          W + (size_t)(nbase + row) * K + k0 + c4 * 4);
      u32x2 st; st.x = pack2(wv.x, wv.y); st.y = pack2(wv.z, wv.w);
      *reinterpret_cast<u32x2*>(&Bsh[row * LDS_STRIDE + c4 * 2]) = st;
      if (MODE == 2) {
        const float4 wu = *reinterpret_cast<const float4*>(
            W + (size_t)(N + nbase + row) * K + k0 + c4 * 4);
        u32x2 su; su.x = pack2(wu.x, wu.y); su.y = pack2(wu.z, wu.w);
        *reinterpret_cast<u32x2*>(&Bsh2[row * LDS_STRIDE + c4 * 2]) = su;
      }
    }
    asm volatile("s_wait_asynccnt 0x0" ::: "memory");
    __syncthreads();

    // ---- fragments: two ds_load_b128 each ----
    const int hi = (lane >= 16) ? 4 : 0;
    FragBF fa[4], fb[2], fb2[2];
#pragma unroll
    for (int t = 0; t < 4; t++) {
      int m = wm * 64 + t * 16 + (lane & 15);
      fa[t].q[0] = *reinterpret_cast<const u32x4*>(&Ash[m * LDS_STRIDE + hi]);
      fa[t].q[1] = *reinterpret_cast<const u32x4*>(&Ash[m * LDS_STRIDE + 8 + hi]);
    }
#pragma unroll
    for (int s = 0; s < 2; s++) {
      int n = wn * 32 + s * 16 + (lane & 15);
      fb[s].q[0] = *reinterpret_cast<const u32x4*>(&Bsh[n * LDS_STRIDE + hi]);
      fb[s].q[1] = *reinterpret_cast<const u32x4*>(&Bsh[n * LDS_STRIDE + 8 + hi]);
      if (MODE == 2) {
        fb2[s].q[0] = *reinterpret_cast<const u32x4*>(&Bsh2[n * LDS_STRIDE + hi]);
        fb2[s].q[1] = *reinterpret_cast<const u32x4*>(&Bsh2[n * LDS_STRIDE + 8 + hi]);
      }
    }
#pragma unroll
    for (int t = 0; t < 4; t++)
#pragma unroll
      for (int s = 0; s < 2; s++) {
        acc[t][s] = __builtin_amdgcn_wmma_f32_16x16x32_bf16(
            false, fa[t].v, false, fb[s].v, (short)0, acc[t][s], false, false);
        if (MODE == 2)
          acc2[t][s] = __builtin_amdgcn_wmma_f32_16x16x32_bf16(
              false, fa[t].v, false, fb2[s].v, (short)0, acc2[t][s], false, false);
      }
    __syncthreads();
  }

  // ---- epilogue: uniform branch; per-row checks only in the one partial tile ----
  if (!GUARD || (mbase + 128 <= NTOK)) {
    gemm_store<MODE, N, false>(acc, acc2, outF, outBF, resid, mbase, nbase, wm, wn, lane);
  } else {
    gemm_store<MODE, N, true>(acc, acc2, outF, outBF, resid, mbase, nbase, wm, wn, lane);
  }
}

// ---------------- prefill flash attention (one wave per (q,h)) ----------------
__global__ __launch_bounds__(256) void prefill_attn_kernel(
    const float* __restrict__ qkv, unsigned short* __restrict__ attn,
    const int* __restrict__ cu, float scale) {
  int lane = threadIdx.x & 31;
  int pair = blockIdx.x * 8 + (threadIdx.x >> 5);
  if (pair >= TPF * NHEAD) return;
  int q = pair >> 5;   // NHEAD==32
  int h = pair & 31;
  int start = 0;
#pragma unroll
  for (int i = 0; i < PB_; i++) {
    if (q >= cu[i + 1]) start = cu[i + 1];
  }
  const float* qp = qkv + (size_t)q * QKV_LD + h * HDSZ + lane * 4;
  float4 qv = *reinterpret_cast<const float4*>(qp);
  qv.x *= scale; qv.y *= scale; qv.z *= scale; qv.w *= scale;

  float mrun = NEGV, lrun = 0.f;
  float4 o = {0.f, 0.f, 0.f, 0.f};

  for (int kb = start; kb <= q; kb += 32) {
    int j = kb + lane;
    bool valid = (j <= q);
    int jr = valid ? j : q;
    const float* kp = qkv + (size_t)jr * QKV_LD + HDIM + h * HDSZ;
    float s = 0.f;
#pragma unroll 8
    for (int c = 0; c < 32; c++) {
      float4 kv = *reinterpret_cast<const float4*>(kp + 4 * c);
      s += __shfl(qv.x, c, 32) * kv.x + __shfl(qv.y, c, 32) * kv.y +
           __shfl(qv.z, c, 32) * kv.z + __shfl(qv.w, c, 32) * kv.w;
    }
    if (!valid) s = NEGV;
    float mn = fmaxf(mrun, wred_max(s));
    float alpha = __expf(mrun - mn);
    float p = __expf(s - mn);
    lrun = lrun * alpha + wred_sum(p);
    o.x *= alpha; o.y *= alpha; o.z *= alpha; o.w *= alpha;
    int nvalid = q - kb + 1; if (nvalid > 32) nvalid = 32;
    for (int jj = 0; jj < nvalid; jj++) {
      float pj = __shfl(p, jj, 32);
      const float* vp =
          qkv + (size_t)(kb + jj) * QKV_LD + 2 * HDIM + h * HDSZ + lane * 4;
      float4 vv = *reinterpret_cast<const float4*>(vp);
      o.x += pj * vv.x; o.y += pj * vv.y; o.z += pj * vv.z; o.w += pj * vv.w;
    }
    mrun = mn;
  }
  float inv = 1.f / lrun;
  uint2 st;
  st.x = pack2(o.x * inv, o.y * inv);
  st.y = pack2(o.z * inv, o.w * inv);
  *reinterpret_cast<uint2*>(attn + (size_t)q * HDIM + h * HDSZ + lane * 4) = st;
}

// ---------------- decode paged attention (one wave per (b,h)) ----------------
__global__ __launch_bounds__(256) void decode_attn_kernel(
    const float* __restrict__ qkv, const float* __restrict__ kheap,
    const float* __restrict__ vheap, const int* __restrict__ btable,
    const int* __restrict__ ctxlens, const int* __restrict__ slotmap,
    unsigned short* __restrict__ attn, float scale) {
  int lane = threadIdx.x & 31;
  int pair = blockIdx.x * 8 + (threadIdx.x >> 5);
  if (pair >= NBD * NHEAD) return;
  int b = pair >> 5;
  int h = pair & 31;
  int ctx = ctxlens[b];
  int dslot = slotmap[TPF + b];   // slot freshly written by this decode token
  int qrow = TPF + b;

  const float* qp = qkv + (size_t)qrow * QKV_LD + h * HDSZ + lane * 4;
  float4 qv = *reinterpret_cast<const float4*>(qp);
  qv.x *= scale; qv.y *= scale; qv.z *= scale; qv.w *= scale;

  float mrun = NEGV, lrun = 0.f;
  float4 o = {0.f, 0.f, 0.f, 0.f};

  for (int base = 0; base < ctx; base += 32) {
    int l = base + lane;
    bool valid = (l < ctx);
    int lc = valid ? l : (ctx - 1);
    int blk = btable[b * MAXB_ + (lc >> 4)];
    int flat = blk * BLOCK_ + (lc & 15);
    const float* kp = (flat == dslot)
        ? (qkv + (size_t)qrow * QKV_LD + HDIM + h * HDSZ)
        : (kheap + (size_t)flat * HDIM + h * HDSZ);
    float s = 0.f;
#pragma unroll 8
    for (int c = 0; c < 32; c++) {
      float4 kv = *reinterpret_cast<const float4*>(kp + 4 * c);
      s += __shfl(qv.x, c, 32) * kv.x + __shfl(qv.y, c, 32) * kv.y +
           __shfl(qv.z, c, 32) * kv.z + __shfl(qv.w, c, 32) * kv.w;
    }
    if (!valid) s = NEGV;
    float mn = fmaxf(mrun, wred_max(s));
    float alpha = __expf(mrun - mn);
    float p = __expf(s - mn);
    lrun = lrun * alpha + wred_sum(p);
    o.x *= alpha; o.y *= alpha; o.z *= alpha; o.w *= alpha;
    int nvalid = ctx - base; if (nvalid > 32) nvalid = 32;
    for (int jj = 0; jj < nvalid; jj++) {
      float pj = __shfl(p, jj, 32);
      int l2 = base + jj;
      int blk2 = btable[b * MAXB_ + (l2 >> 4)];
      int flat2 = blk2 * BLOCK_ + (l2 & 15);
      const float* vp = (flat2 == dslot)
          ? (qkv + (size_t)qrow * QKV_LD + 2 * HDIM + h * HDSZ)
          : (vheap + (size_t)flat2 * HDIM + h * HDSZ);
      float4 vv = *reinterpret_cast<const float4*>(vp + lane * 4);
      o.x += pj * vv.x; o.y += pj * vv.y; o.z += pj * vv.z; o.w += pj * vv.w;
    }
    mrun = mn;
  }
  float inv = 1.f / lrun;
  uint2 st;
  st.x = pack2(o.x * inv, o.y * inv);
  st.y = pack2(o.z * inv, o.w * inv);
  *reinterpret_cast<uint2*>(attn + (size_t)qrow * HDIM + h * HDSZ + lane * 4) = st;
}

// ---------------- driver ----------------
extern "C" void kernel_launch(void* const* d_in, const int* in_sizes, int n_in,
                              void* d_out, int out_size, void* d_ws, size_t ws_size,
                              hipStream_t stream) {
  (void)in_sizes; (void)n_in; (void)out_size; (void)ws_size;
  const float* x       = (const float*)d_in[0];
  const float* kheap   = (const float*)d_in[1];
  const float* vheap   = (const float*)d_in[2];
  const int*   slotmap = (const int*)d_in[3];
  const int*   btable  = (const int*)d_in[4];
  const int*   ctx     = (const int*)d_in[5];
  const int*   cu      = (const int*)d_in[6];
  const float* norm1_w = (const float*)d_in[10];
  const float* norm2_w = (const float*)d_in[11];
  const float* w_qkv   = (const float*)d_in[12];
  const float* wo      = (const float*)d_in[13];
  const float* w_gu    = (const float*)d_in[14];
  const float* w2      = (const float*)d_in[15];
  float* out = (float*)d_out;

  char* wsp = (char*)d_ws;
  auto carve = [&](size_t bytes) {
    char* p = wsp;
    wsp += (bytes + 255) & ~(size_t)255;
    return p;
  };
  // buffers written by unguarded GEMM stores are padded to MPAD rows
  unsigned short* xn   = (unsigned short*)carve((size_t)NTOK * DIM * 2);
  float*          qkv  = (float*)carve((size_t)MPAD * QKV_LD * 4);
  unsigned short* attn = (unsigned short*)carve((size_t)NTOK * HDIM * 2);
  float*          hbuf = (float*)carve((size_t)MPAD * DIM * 4);
  unsigned short* hn   = (unsigned short*)carve((size_t)NTOK * DIM * 2);
  unsigned short* act  = (unsigned short*)carve((size_t)MPAD * FF * 2);

  const float scale = 0.08838834764831845f;  // 1/sqrt(128)

  rmsnorm_bf16_kernel<<<NTOK, 256, 0, stream>>>(x, norm1_w, xn);
  gemm_wmma_kernel<0, QKV_LD, DIM, false><<<dim3(MTILES, QKV_LD / 128), 256, 0, stream>>>(
      xn, w_qkv, qkv, nullptr, nullptr);
  prefill_attn_kernel<<<(TPF * NHEAD) / 8, 256, 0, stream>>>(qkv, attn, cu, scale);
  decode_attn_kernel<<<(NBD * NHEAD) / 8, 256, 0, stream>>>(
      qkv, kheap, vheap, btable, ctx, slotmap, attn, scale);
  gemm_wmma_kernel<1, DIM, HDIM, false><<<dim3(MTILES, DIM / 128), 256, 0, stream>>>(
      attn, wo, hbuf, nullptr, x);
  rmsnorm_bf16_kernel<<<NTOK, 256, 0, stream>>>(hbuf, norm2_w, hn);
  gemm_wmma_kernel<2, FF, DIM, false><<<dim3(MTILES, FF / 128), 256, 0, stream>>>(
      hn, w_gu, nullptr, act, nullptr);
  gemm_wmma_kernel<1, DIM, FF, true><<<dim3(MTILES, DIM / 128), 256, 0, stream>>>(
      act, w2, out, nullptr, hbuf);
}
